// GRUTrajectoryDecoder_81114752352795
// MI455X (gfx1250) — compile-verified, matrix operand order checked by reading, not run on hardware
//
#include <hip/hip_runtime.h>

typedef __bf16 bf16_t;
typedef __attribute__((ext_vector_type(16))) __bf16 v16bf;
typedef __attribute__((ext_vector_type(8)))  __bf16 v8bf;
typedef __attribute__((ext_vector_type(8)))  float  v8f;
typedef __attribute__((ext_vector_type(4)))  unsigned int u32x4;
typedef __attribute__((ext_vector_type(8)))  int i32x8;
typedef __attribute__((ext_vector_type(4)))  int i32x4;

#define BS     1024
#define IN_DIM 256
#define H      3000
#define HP     3072
#define NG     (3 * HP)
#define ITEM   3
#define STEPS  30

// LDS staging for B: 128 rows x 64 K bf16, TDM-padded to 72 elements/row
// (32 data DWORDs + 4 pad DWORDs) so ds_load_b128 rows hit distinct banks.
#define LDS_ROW_EL   72
#define LDS_BUF_EL   (128 * LDS_ROW_EL)        // 9216 elements
#define LDS_BUF_B    (LDS_BUF_EL * 2)          // 18432 bytes
#define LDS_TOTAL_B  (2 * LDS_BUF_B)           // 36864 bytes (double buffer)

__device__ __forceinline__ bf16_t f2bf(float f) { return (bf16_t)f; }

// ---- TDM: DMA a 64(K) x 128(row) bf16 tile, row-major src (stride ld_elems),
// into LDS at byte offset lds_off with 4-DWORD pad per 32-DWORD row. ----
__device__ __forceinline__ void tdm_load_tile(unsigned lds_off,
                                              const bf16_t* __restrict__ gsrc,
                                              unsigned ld_elems) {
  unsigned long long ga = (unsigned long long)(size_t)gsrc;
  u32x4 g0;
  g0[0] = 1u;                                            // count=1 (valid D#)
  g0[1] = lds_off;                                       // lds_addr (bytes)
  g0[2] = (unsigned)ga;                                  // global_addr[31:0]
  g0[3] = (unsigned)((ga >> 32) & 0x01ffffffu)           // global_addr[56:32]
        | (2u << 30);                                    // type=2 (image)
  i32x8 g1;
  g1[0] = (int)((1u << 16)        // data_size = 2 bytes
              | (1u << 20)        // pad_enable
              | (4u << 22)        // pad_interval: every 32 DWORDs
              | (3u << 25));      // pad_amount: 4 DWORDs
  g1[1] = (int)(64u << 16);       // tensor_dim0 = 64 (lo16 @ bits 63:48)
  g1[2] = (int)(128u << 16);      // tensor_dim0 hi=0 | tensor_dim1 = 128
  g1[3] = (int)(64u << 16);       // tensor_dim1 hi=0 | tile_dim0 = 64
  g1[4] = (int)(128u);            // tile_dim1 = 128 | tile_dim2 = 0
  g1[5] = (int)ld_elems;          // tensor_dim0_stride lo32 (data_size units)
  g1[6] = 0;                      // stride0 hi | stride1 lo (unused, 2D)
  g1[7] = 0;
  i32x4 z4;
  z4[0] = 0; z4[1] = 0; z4[2] = 0; z4[3] = 0;
  i32x8 z8;
  z8[0] = 0; z8[1] = 0; z8[2] = 0; z8[3] = 0;
  z8[4] = 0; z8[5] = 0; z8[6] = 0; z8[7] = 0;
  __builtin_amdgcn_tensor_load_to_lds(g0, g1, z4, z4, z8, 0);
}

// Fragment of A from global: 16(M/N) x 32(K), row-major, CDNA5 16-bit layout.
__device__ __forceinline__ v16bf load_frag(const bf16_t* __restrict__ base,
                                           int ld, int row0, int k, int lane) {
  const int r  = row0 + (lane & 15);
  const int ko = (lane >> 4) << 3;
  const bf16_t* p = base + (size_t)r * ld + (size_t)(k + ko);
  v8bf lo = *(const v8bf*)(p);
  v8bf hi = *(const v8bf*)(p + 16);
  v16bf out;
#pragma unroll
  for (int i = 0; i < 8; ++i) { out[i] = lo[i]; out[i + 8] = hi[i]; }
  return out;
}

// Fragment of B from the padded LDS tile (row stride 72 elements).
__device__ __forceinline__ v16bf lds_frag(const bf16_t* __restrict__ sB,
                                          int row0, int kk, int lane) {
  const int r  = row0 + (lane & 15);
  const int ko = (lane >> 4) << 3;
  const bf16_t* p = sB + r * LDS_ROW_EL + kk * 32 + ko;
  v8bf lo = *(const v8bf*)(p);
  v8bf hi = *(const v8bf*)(p + 16);
  v16bf out;
#pragma unroll
  for (int i = 0; i < 8; ++i) { out[i] = lo[i]; out[i + 8] = hi[i]; }
  return out;
}

// C[M,N] = act(A[M,K] * B[N,K]^T + bias[N]); dims exact multiples of tiles.
// Block: 256 threads = 8 waves (4 in M x 2 in N); block tile 128x128.
// B tile TDM-staged to LDS (double-buffered); A direct from global/L2.
__global__ __launch_bounds__(256) void k_gemm_bf16(
    const bf16_t* __restrict__ A, const bf16_t* __restrict__ B,
    const float* __restrict__ bias,
    float* __restrict__ outF, bf16_t* __restrict__ outB,
    int K, int ldA, int ldB, int ldOF, int ldOB, int relu)
{
  extern __shared__ char dynsm[];
  bf16_t* smB = (bf16_t*)dynsm;

  const int lane = threadIdx.x & 31;
  const int wave = threadIdx.x >> 5;
  const int wm = wave & 3, wn = wave >> 2;
  const int row0 = blockIdx.x * 128 + wm * 32;
  const int col0 = blockIdx.y * 128;           // block's B-row base

  // Wave-uniform scalar predicate: forces s_cmp/s_cbranch so waves 1..7
  // truly skip the TDM block (TDM issues regardless of EXEC otherwise).
  const int swave = __builtin_amdgcn_readfirstlane(threadIdx.x) >> 5;
  const bool producer = (swave == 0);

  v8f acc[2][4];
#pragma unroll
  for (int i = 0; i < 2; ++i)
#pragma unroll
    for (int j = 0; j < 4; ++j)
#pragma unroll
      for (int e = 0; e < 8; ++e) acc[i][j][e] = 0.0f;

  const int nchunk = K >> 6;                   // K / 64
  if (producer)
    tdm_load_tile(0u, B + (size_t)col0 * ldB, (unsigned)ldB);

  int cur = 0;
  for (int kc = 0; kc < nchunk; ++kc) {
    if (producer)
      __builtin_amdgcn_s_wait_tensorcnt((short)0);   // current buffer landed
    __syncthreads();                                 // ...and prev buffer free
    if (producer && kc + 1 < nchunk)
      tdm_load_tile((unsigned)((cur ^ 1) * LDS_BUF_B),
                    B + (size_t)col0 * ldB + (size_t)(kc + 1) * 64,
                    (unsigned)ldB);                  // overlaps compute below

    const bf16_t* sB = smB + cur * LDS_BUF_EL;
    const int kbase = kc * 64;
#pragma unroll
    for (int kk = 0; kk < 2; ++kk) {
      v16bf a0 = load_frag(A, ldA, row0,      kbase + kk * 32, lane);
      v16bf a1 = load_frag(A, ldA, row0 + 16, kbase + kk * 32, lane);
      v16bf bb[4];
#pragma unroll
      for (int t = 0; t < 4; ++t)
        bb[t] = lds_frag(sB, wn * 64 + 16 * t, kk, lane);
#pragma unroll
      for (int t = 0; t < 4; ++t) {
        acc[0][t] = __builtin_amdgcn_wmma_f32_16x16x32_bf16(
            false, a0, false, bb[t], (short)0, acc[0][t], false, false);
        acc[1][t] = __builtin_amdgcn_wmma_f32_16x16x32_bf16(
            false, a1, false, bb[t], (short)0, acc[1][t], false, false);
      }
    }
    cur ^= 1;
  }

  const int nlane = lane & 15;
  const int mbase = (lane >> 4) << 3;
#pragma unroll
  for (int mi = 0; mi < 2; ++mi) {
#pragma unroll
    for (int t = 0; t < 4; ++t) {
      const int col = blockIdx.y * 128 + wn * 64 + 16 * t + nlane;
      const float bv = bias ? bias[col] : 0.0f;
#pragma unroll
      for (int r = 0; r < 8; ++r) {
        const int row = row0 + mi * 16 + mbase + r;
        float v = acc[mi][t][r] + bv;
        if (relu) v = v > 0.0f ? v : 0.0f;
        if (outF) outF[(size_t)row * ldOF + col] = v;
        if (outB) outB[(size_t)row * ldOB + col] = f2bf(v);
      }
    }
  }
}

__global__ void k_f2b(const float* __restrict__ s, bf16_t* __restrict__ d, int n) {
  int i = blockIdx.x * blockDim.x + threadIdx.x;
  if (i < n) d[i] = f2bf(s[i]);
}

__global__ void k_pad2d_f2b(const float* __restrict__ s, bf16_t* __restrict__ d,
                            int srcR, int srcC, int dstC, int total) {
  int i = blockIdx.x * blockDim.x + threadIdx.x;
  if (i >= total) return;
  int r = i / dstC, c = i % dstC;
  float v = (r < srcR && c < srcC) ? s[(size_t)r * srcC + c] : 0.0f;
  d[i] = f2bf(v);
}

__global__ void k_pad_whh(const float* __restrict__ s, bf16_t* __restrict__ d) {
  int i = blockIdx.x * blockDim.x + threadIdx.x;
  if (i >= NG * HP) return;
  int R = i / HP, k = i % HP;
  int g = R / HP, ii = R % HP;
  float v = (ii < H && k < H) ? s[((size_t)g * H + ii) * H + k] : 0.0f;
  d[i] = f2bf(v);
}

__global__ void k_pad_bias_gate(const float* __restrict__ s, float* __restrict__ d) {
  int i = blockIdx.x * blockDim.x + threadIdx.x;
  if (i >= NG) return;
  int g = i / HP, r = i % HP;
  d[i] = (r < H) ? s[g * H + r] : 0.0f;
}

__global__ void k_pad_bias(const float* __restrict__ s, float* __restrict__ d,
                           int srcN, int dstN) {
  int i = blockIdx.x * blockDim.x + threadIdx.x;
  if (i >= dstN) return;
  d[i] = (i < srcN) ? s[i] : 0.0f;
}

__global__ void k_zero_f(float* __restrict__ d, int n) {
  int i = blockIdx.x * blockDim.x + threadIdx.x;
  if (i < n) d[i] = 0.0f;
}

__global__ __launch_bounds__(256) void k_gate(
    const float* __restrict__ gh, const float* __restrict__ xb,
    const float* __restrict__ Wih, const float* __restrict__ bih,
    float* __restrict__ hF, bf16_t* __restrict__ hB)
{
  int i = blockIdx.x * blockDim.x + threadIdx.x;
  int b = i / HP, j = i % HP;
  float hn = 0.0f;
  if (j < H) {
    const float x0 = xb[b * 3 + 0], x1 = xb[b * 3 + 1], x2 = xb[b * 3 + 2];
    const float ir  = Wih[(size_t)j * 3 + 0] * x0 + Wih[(size_t)j * 3 + 1] * x1 +
                      Wih[(size_t)j * 3 + 2] * x2 + bih[j];
    const float iz  = Wih[(size_t)(H + j) * 3 + 0] * x0 + Wih[(size_t)(H + j) * 3 + 1] * x1 +
                      Wih[(size_t)(H + j) * 3 + 2] * x2 + bih[H + j];
    const float in_ = Wih[(size_t)(2 * H + j) * 3 + 0] * x0 + Wih[(size_t)(2 * H + j) * 3 + 1] * x1 +
                      Wih[(size_t)(2 * H + j) * 3 + 2] * x2 + bih[2 * H + j];
    const float* g = gh + (size_t)b * NG;
    const float r = 1.0f / (1.0f + __expf(-(ir + g[j])));
    const float u = 1.0f / (1.0f + __expf(-(iz + g[HP + j])));
    const float n = tanhf(in_ + r * g[2 * HP + j]);
    const float ho = hF[(size_t)b * HP + j];
    hn = (1.0f - u) * n + u * ho;
  }
  hF[(size_t)b * HP + j] = hn;
  hB[(size_t)b * HP + j] = f2bf(hn);
}

__global__ __launch_bounds__(256) void k_readout(
    const float* __restrict__ hF, const float* __restrict__ Wo,
    const float* __restrict__ bo, float* __restrict__ xb,
    float* __restrict__ out, int step)
{
  __shared__ float sm[3][256];
  const int b = blockIdx.x, t = threadIdx.x;
  float s0 = 0.f, s1 = 0.f, s2 = 0.f;
  const float* hp = hF + (size_t)b * HP;
  for (int j = t; j < H; j += 256) {
    float hv = hp[j];
    s0 += hv * Wo[j];
    s1 += hv * Wo[H + j];
    s2 += hv * Wo[2 * H + j];
  }
  sm[0][t] = s0; sm[1][t] = s1; sm[2][t] = s2;
  __syncthreads();
  for (int o = 128; o > 0; o >>= 1) {
    if (t < o) {
      sm[0][t] += sm[0][t + o];
      sm[1][t] += sm[1][t + o];
      sm[2][t] += sm[2][t + o];
    }
    __syncthreads();
  }
  if (t < 3) {
    float xn = sm[t][0] + bo[t] + xb[b * 3 + t];
    out[((size_t)b * STEPS + step) * 3 + t] = xn;
    xb[b * 3 + t] = xn;
  }
}

extern "C" void kernel_launch(void* const* d_in, const int* in_sizes, int n_in,
                              void* d_out, int out_size, void* d_ws, size_t ws_size,
                              hipStream_t stream) {
  const float* z   = (const float*)d_in[0];
  const float* W1  = (const float*)d_in[1];
  const float* b1  = (const float*)d_in[2];
  const float* W2  = (const float*)d_in[3];
  const float* b2  = (const float*)d_in[4];
  const float* W3  = (const float*)d_in[5];
  const float* b3  = (const float*)d_in[6];
  const float* Wih = (const float*)d_in[7];
  const float* bih = (const float*)d_in[8];
  const float* Whh = (const float*)d_in[9];
  const float* bhh = (const float*)d_in[10];
  const float* Wo  = (const float*)d_in[11];
  const float* bo  = (const float*)d_in[12];
  float* out = (float*)d_out;
  (void)in_sizes; (void)n_in; (void)out_size; (void)ws_size;

  char* ws = (char*)d_ws;
  size_t off = 0;
  auto alloc = [&](size_t bytes) -> char* {
    off = (off + 255) & ~(size_t)255;
    char* p = ws + off;
    off += bytes;
    return p;
  };
  bf16_t* zB   = (bf16_t*)alloc((size_t)BS * IN_DIM * 2);
  bf16_t* W1b  = (bf16_t*)alloc((size_t)512 * IN_DIM * 2);
  bf16_t* W2b  = (bf16_t*)alloc((size_t)256 * 512 * 2);
  bf16_t* W3b  = (bf16_t*)alloc((size_t)HP * 256 * 2);
  float*  b3p  = (float*)alloc((size_t)HP * 4);
  bf16_t* WhhB = (bf16_t*)alloc((size_t)NG * HP * 2);   // ~56.6 MB, L2-resident
  float*  bhhP = (float*)alloc((size_t)NG * 4);
  bf16_t* act1 = (bf16_t*)alloc((size_t)BS * 512 * 2);
  bf16_t* act2 = (bf16_t*)alloc((size_t)BS * 256 * 2);
  float*  hF   = (float*)alloc((size_t)BS * HP * 4);
  bf16_t* hB   = (bf16_t*)alloc((size_t)BS * HP * 2);
  float*  gh   = (float*)alloc((size_t)BS * NG * 4);
  float*  xb   = (float*)alloc((size_t)BS * ITEM * 4);

  const int T = 256;
  k_f2b<<<(BS * IN_DIM + T - 1) / T, T, 0, stream>>>(z, zB, BS * IN_DIM);
  k_f2b<<<(512 * IN_DIM + T - 1) / T, T, 0, stream>>>(W1, W1b, 512 * IN_DIM);
  k_f2b<<<(256 * 512 + T - 1) / T, T, 0, stream>>>(W2, W2b, 256 * 512);
  k_pad2d_f2b<<<(HP * 256 + T - 1) / T, T, 0, stream>>>(W3, W3b, H, 256, 256, HP * 256);
  k_pad_bias<<<(HP + T - 1) / T, T, 0, stream>>>(b3, b3p, H, HP);
  k_pad_whh<<<(NG * HP + T - 1) / T, T, 0, stream>>>(Whh, WhhB);
  k_pad_bias_gate<<<(NG + T - 1) / T, T, 0, stream>>>(bhh, bhhP);
  k_zero_f<<<(BS * ITEM + T - 1) / T, T, 0, stream>>>(xb, BS * ITEM);

  dim3 blk(256);
  k_gemm_bf16<<<dim3(BS / 128, 512 / 128), blk, LDS_TOTAL_B, stream>>>(
      zB, W1b, b1, nullptr, act1, IN_DIM, IN_DIM, IN_DIM, 0, 512, 1);
  k_gemm_bf16<<<dim3(BS / 128, 256 / 128), blk, LDS_TOTAL_B, stream>>>(
      act1, W2b, b2, nullptr, act2, 512, 512, 512, 0, 256, 1);
  k_gemm_bf16<<<dim3(BS / 128, HP / 128), blk, LDS_TOTAL_B, stream>>>(
      act2, W3b, b3p, hF, hB, 256, 256, 256, HP, HP, 1);

  for (int s = 0; s < STEPS; ++s) {
    k_gemm_bf16<<<dim3(BS / 128, NG / 128), blk, LDS_TOTAL_B, stream>>>(
        hB, WhhB, bhhP, gh, nullptr, HP, HP, HP, NG, 0, 0);
    k_gate<<<(BS * HP) / T, T, 0, stream>>>(gh, xb, Wih, bih, hF, hB);
    k_readout<<<BS, T, 0, stream>>>(hF, Wo, bo, xb, out, s);
  }
}
